// LinearAttention_5669356834836
// MI455X (gfx1250) — compile-verified, matrix operand order checked by reading, not run on hardware
//
#include <hip/hip_runtime.h>
#include <hip/hip_bf16.h>
#include <stdint.h>

// ---------------------------------------------------------------------------
// Problem constants (B,C,H,W)=(16,256,64,64), HEADS=8, DIM_HEAD=64
// ---------------------------------------------------------------------------
#define BB    16
#define CC    256
#define NN    4096        // H*W
#define HID   512         // HEADS*DIM_HEAD
#define OQKV  1536        // 3*HID
#define NH    8
#define DH    64

typedef __attribute__((ext_vector_type(16))) __bf16 v16bf;
typedef __attribute__((ext_vector_type(8)))  float  v8f;

struct alignas(16) U4 { unsigned x, y, z, w; };

union Frag { v16bf bf; U4 q[2]; };   // 16 bf16 = one WMMA A/B operand per lane

// fp32 -> bf16 using hardware converts (v_cvt_pk_bf16_f32 when available)
static __device__ __forceinline__ unsigned pk(float a, float b) {
#if __has_builtin(__builtin_amdgcn_cvt_pk_bf16_f32)
  typedef __attribute__((ext_vector_type(2))) __bf16 v2bf;
  v2bf p = __builtin_amdgcn_cvt_pk_bf16_f32(a, b);
  return __builtin_bit_cast(unsigned, p);
#else
  unsigned short lo = __builtin_bit_cast(unsigned short, (__bf16)a);
  unsigned short hi = __builtin_bit_cast(unsigned short, (__bf16)b);
  return (unsigned)lo | ((unsigned)hi << 16);
#endif
}
static __device__ __forceinline__ unsigned short f2bf(float f) {
  return __builtin_bit_cast(unsigned short, (__bf16)f);
}
static __device__ __forceinline__ float bf2f(unsigned short s) {
  return __builtin_bit_cast(float, (unsigned)s << 16);
}

static __device__ __forceinline__ v8f wmma_bf16(const Frag& a, const Frag& b, v8f c) {
  return __builtin_amdgcn_wmma_f32_16x16x32_bf16(
      /*neg_a=*/false, a.bf, /*neg_b=*/false, b.bf,
      /*c_mod=*/(short)0, c, /*reuse_a=*/false, /*reuse_b=*/false);
}

// Async global->LDS 16-byte copy (CDNA5 GLOBAL_LOAD_ASYNC_TO_LDS_B128).
// ldsOff = low 32 bits of a generic LDS pointer (== LDS byte offset per ISA 10.2).
static __device__ __forceinline__ void async_g2l_b128(unsigned ldsOff, const void* gsrc) {
  unsigned long long ga = (unsigned long long)gsrc;
  asm volatile("global_load_async_to_lds_b128 %0, %1, off"
               :: "v"(ldsOff), "v"(ga) : "memory");
}
static __device__ __forceinline__ void wait_asynccnt0() {
  asm volatile("s_wait_asynccnt 0x0" ::: "memory");
}

// ---------------------------------------------------------------------------
// Kernel 1: qkv[b, o, n] = sum_c w_qkv[o,c] * x[b,c,n]   (o<512 pre-scaled 1/8)
// 128x128 tile per workgroup, K stepped by 32, bf16 WMMA, f32 accum.
// LDS operand tiles are stored in WMMA fragment order:
//   fragment(lane L, elem j): row = tile*16 + (L&15),
//   k = ((j>>3)*16) + ((L>>4)*8) + (j&7)  -> two contiguous 8-elem runs.
// ---------------------------------------------------------------------------
__global__ __launch_bounds__(256)
void k_qkv(const float* __restrict__ x, const float* __restrict__ wqkv,
           unsigned short* __restrict__ qkvb) {
  __shared__ unsigned short lA[8 * 32 * 16];   // 8 KB
  __shared__ unsigned short lB[8 * 32 * 16];   // 8 KB
  const int b  = blockIdx.z;
  const int o0 = blockIdx.y * 128;
  const int n0 = blockIdx.x * 128;
  const int t = threadIdx.x, lane = t & 31, w = t >> 5;
  const int wr = w >> 1, wc = w & 1;

  v8f acc[2][4];
#pragma unroll
  for (int i = 0; i < 2; ++i)
#pragma unroll
    for (int j = 0; j < 4; ++j) acc[i][j] = 0.f;

  for (int ks = 0; ks < CC; ks += 32) {
    __syncthreads();
    // ---- stage A tile: w_qkv[o0..+128, ks..+32] fp32 -> bf16 fragments
#pragma unroll
    for (int p = 0; p < 2; ++p) {
      int r  = t + p * 256;           // 512 runs of 8
      int m  = r >> 2;
      int k0 = (r & 3) * 8;
      const float* src = wqkv + (size_t)(o0 + m) * CC + ks + k0;
      if (ks + 32 < CC) __builtin_prefetch(src + 32, 0, 3);
      float4 f0 = *(const float4*)src;
      float4 f1 = *(const float4*)(src + 4);
      U4 o; o.x = pk(f0.x, f0.y); o.y = pk(f0.z, f0.w);
            o.z = pk(f1.x, f1.y); o.w = pk(f1.z, f1.w);
      int ln = (m & 15) | ((k0 & 8) ? 16 : 0);
      *(U4*)&lA[(((m >> 4) * 32 + ln) * 16) + ((k0 & 16) ? 8 : 0)] = o;
    }
    // ---- stage B tile: x[b, ks..+32, n0..+128] fp32 -> bf16 fragments (N-major)
#pragma unroll
    for (int p = 0; p < 2; ++p) {
      int r  = t + p * 256;
      int n  = r & 127;
      int k0 = (r >> 7) * 8;
      const float* src = x + ((size_t)b * CC + ks + k0) * NN + n0 + n;
      float f[8];
#pragma unroll
      for (int i = 0; i < 8; ++i) f[i] = src[(size_t)i * NN];
      U4 o; o.x = pk(f[0], f[1]); o.y = pk(f[2], f[3]);
            o.z = pk(f[4], f[5]); o.w = pk(f[6], f[7]);
      int ln = (n & 15) | ((k0 & 8) ? 16 : 0);
      *(U4*)&lB[(((n >> 4) * 32 + ln) * 16) + ((k0 & 16) ? 8 : 0)] = o;
    }
    __syncthreads();
    // ---- compute: wave = 32(M) x 64(N) -> 8 WMMAs per K-step
    Frag af[2];
#pragma unroll
    for (int mi = 0; mi < 2; ++mi) {
      const U4* ap = (const U4*)&lA[((wr * 2 + mi) * 32 + lane) * 16];
      af[mi].q[0] = ap[0]; af[mi].q[1] = ap[1];
    }
#pragma unroll
    for (int ni = 0; ni < 4; ++ni) {
      Frag bf_;
      const U4* bp = (const U4*)&lB[((wc * 4 + ni) * 32 + lane) * 16];
      bf_.q[0] = bp[0]; bf_.q[1] = bp[1];
#pragma unroll
      for (int mi = 0; mi < 2; ++mi) acc[mi][ni] = wmma_bf16(af[mi], bf_, acc[mi][ni]);
    }
  }
  // ---- store (bf16), scale q rows by DIM_HEAD^-0.5 = 1/8
#pragma unroll
  for (int mi = 0; mi < 2; ++mi)
#pragma unroll
    for (int ni = 0; ni < 4; ++ni) {
      int ncol = n0 + wc * 64 + ni * 16 + (lane & 15);
#pragma unroll
      for (int r = 0; r < 8; ++r) {
        int orow = o0 + wr * 32 + mi * 16 + r + ((lane >> 4) * 8);
        float v = acc[mi][ni][r];
        if (orow < HID) v *= 0.125f;
        qkvb[((size_t)b * OQKV + orow) * NN + ncol] = f2bf(v);
      }
    }
}

// ---------------------------------------------------------------------------
// Kernel 2: per-row (b, hd) max and sum(exp) over n for k rows
// ---------------------------------------------------------------------------
__global__ __launch_bounds__(256)
void k_stats(const unsigned short* __restrict__ qkvb,
             float* __restrict__ rowmax, float* __restrict__ rowsum) {
  const int row = blockIdx.x;            // b*512 + hd
  const int b = row >> 9, hd = row & 511;
  const unsigned short* src = qkvb + ((size_t)b * OQKV + HID + hd) * NN;
  const int t = threadIdx.x;

  float vals[16], m = -3.0e38f;
#pragma unroll
  for (int i = 0; i < 16; ++i) { vals[i] = bf2f(src[t + i * 256]); m = fmaxf(m, vals[i]); }
#pragma unroll
  for (int off = 16; off >= 1; off >>= 1) m = fmaxf(m, __shfl_xor(m, off, 32));
  __shared__ float red[8];
  if ((t & 31) == 0) red[t >> 5] = m;
  __syncthreads();
  float mm = red[0];
#pragma unroll
  for (int j = 1; j < 8; ++j) mm = fmaxf(mm, red[j]);

  float s = 0.f;
#pragma unroll
  for (int i = 0; i < 16; ++i) s += __expf(vals[i] - mm);
#pragma unroll
  for (int off = 16; off >= 1; off >>= 1) s += __shfl_xor(s, off, 32);
  __syncthreads();
  if ((t & 31) == 0) red[t >> 5] = s;
  __syncthreads();
  if (t == 0) {
    float ss = 0.f;
#pragma unroll
    for (int j = 0; j < 8; ++j) ss += red[j];
    rowmax[row] = mm;
    rowsum[row] = ss;
  }
}

// ---------------------------------------------------------------------------
// Kernel 3: ctx[b,h,d,e] = (sum_n exp(k[d,n]-max_d) * v[e,n]) / sum_d
// One workgroup (4 waves) per (b,h); wave w owns d-subtile w, all 4 e-subtiles.
// exp applied in-register during A-fragment load.
// ---------------------------------------------------------------------------
static __device__ __forceinline__ unsigned expPair(unsigned p, float rm) {
  float lo = bf2f((unsigned short)(p & 0xffffu));
  float hi = bf2f((unsigned short)(p >> 16));
  return pk(__expf(lo - rm), __expf(hi - rm));
}
static __device__ __forceinline__ U4 exp8(U4 v, float rm) {
  U4 r; r.x = expPair(v.x, rm); r.y = expPair(v.y, rm);
        r.z = expPair(v.z, rm); r.w = expPair(v.w, rm);
  return r;
}

__global__ __launch_bounds__(128)
void k_ctx(const unsigned short* __restrict__ qkvb,
           const float* __restrict__ rowmax, const float* __restrict__ rowsum,
           float* __restrict__ ctx) {
  const int bh = blockIdx.x;             // b*8 + h
  const int b = bh >> 3, h = bh & 7;
  const int t = threadIdx.x, lane = t & 31, w = t >> 5;   // w = d-subtile
  const unsigned short* kb = qkvb + ((size_t)b * OQKV + HID + h * DH) * NN;
  const unsigned short* vb = qkvb + ((size_t)b * OQKV + 2 * HID + h * DH) * NN;

  const int drow  = w * 16 + (lane & 15);
  const float rm  = rowmax[b * HID + h * DH + drow];
  const int base0 = (lane >> 4) * 8;

  v8f acc[4];
#pragma unroll
  for (int i = 0; i < 4; ++i) acc[i] = 0.f;

  for (int ns = 0; ns < NN; ns += 32) {
    Frag a;
    const unsigned short* ksrc = kb + (size_t)drow * NN + ns + base0;
    a.q[0] = exp8(*(const U4*)ksrc, rm);
    a.q[1] = exp8(*(const U4*)(ksrc + 16), rm);
#pragma unroll
    for (int ni = 0; ni < 4; ++ni) {
      int e = ni * 16 + (lane & 15);
      const unsigned short* vsrc = vb + (size_t)e * NN + ns + base0;
      Frag bf_;
      bf_.q[0] = *(const U4*)vsrc;
      bf_.q[1] = *(const U4*)(vsrc + 16);
      acc[ni] = wmma_bf16(a, bf_, acc[ni]);
    }
  }
#pragma unroll
  for (int ni = 0; ni < 4; ++ni)
#pragma unroll
    for (int r = 0; r < 8; ++r) {
      int dl = w * 16 + r + ((lane >> 4) * 8);
      int e  = ni * 16 + (lane & 15);
      ctx[((size_t)bh * DH + dl) * DH + e] = acc[ni][r] / rowsum[b * HID + h * DH + dl];
    }
}

// ---------------------------------------------------------------------------
// Kernel 4: M[b][o][h*64+d] = sum_e w_out[o, h*64+e] * ctx[b,h,d,e]  (bf16 out)
// One workgroup (8 waves) per (h, b). Tiny GEMM 256x64x64; direct global loads.
// ---------------------------------------------------------------------------
__global__ __launch_bounds__(256)
void k_fold(const float* __restrict__ wout, const float* __restrict__ ctx,
            unsigned short* __restrict__ Mb) {
  const int h = blockIdx.x, b = blockIdx.y;
  const int t = threadIdx.x, lane = t & 31, w = t >> 5;
  const int base0 = (lane >> 4) * 8;

  v8f acc[2][4];
#pragma unroll
  for (int i = 0; i < 2; ++i)
#pragma unroll
    for (int j = 0; j < 4; ++j) acc[i][j] = 0.f;

  for (int ks = 0; ks < DH; ks += 32) {
    Frag af[2];
#pragma unroll
    for (int mi = 0; mi < 2; ++mi) {
      int orow = (w * 2 + mi) * 16 + (lane & 15);
      const float* s = wout + (size_t)orow * HID + h * DH + ks + base0;
      float4 f0 = *(const float4*)s, f1 = *(const float4*)(s + 4);
      af[mi].q[0].x = pk(f0.x, f0.y); af[mi].q[0].y = pk(f0.z, f0.w);
      af[mi].q[0].z = pk(f1.x, f1.y); af[mi].q[0].w = pk(f1.z, f1.w);
      // second 8-run: k = 16 + base0 .. +7
      const float* s2 = s + 16;
      float4 g0 = *(const float4*)s2, g1 = *(const float4*)(s2 + 4);
      af[mi].q[1].x = pk(g0.x, g0.y); af[mi].q[1].y = pk(g0.z, g0.w);
      af[mi].q[1].z = pk(g1.x, g1.y); af[mi].q[1].w = pk(g1.z, g1.w);
    }
#pragma unroll
    for (int ni = 0; ni < 4; ++ni) {
      int d = ni * 16 + (lane & 15);
      const float* s = ctx + ((size_t)(b * NH + h) * DH + d) * DH + ks + base0;
      Frag bf_;
      float4 f0 = *(const float4*)s, f1 = *(const float4*)(s + 4);
      bf_.q[0].x = pk(f0.x, f0.y); bf_.q[0].y = pk(f0.z, f0.w);
      bf_.q[0].z = pk(f1.x, f1.y); bf_.q[0].w = pk(f1.z, f1.w);
      const float* s2 = s + 16;
      float4 g0 = *(const float4*)s2, g1 = *(const float4*)(s2 + 4);
      bf_.q[1].x = pk(g0.x, g0.y); bf_.q[1].y = pk(g0.z, g0.w);
      bf_.q[1].z = pk(g1.x, g1.y); bf_.q[1].w = pk(g1.z, g1.w);
#pragma unroll
      for (int mi = 0; mi < 2; ++mi) acc[mi][ni] = wmma_bf16(af[mi], bf_, acc[mi][ni]);
    }
  }
#pragma unroll
  for (int mi = 0; mi < 2; ++mi)
#pragma unroll
    for (int ni = 0; ni < 4; ++ni)
#pragma unroll
      for (int r = 0; r < 8; ++r) {
        int orow = (w * 2 + mi) * 16 + r + ((lane >> 4) * 8);
        int d    = ni * 16 + (lane & 15);
        Mb[((size_t)b * CC + orow) * HID + h * DH + d] = f2bf(acc[mi][ni][r]);
      }
}

// ---------------------------------------------------------------------------
// Kernel 5: y[b][o][n] = sum_hd M[b][o][hd] * q[b][hd][n] + b_out[o]  (fp32 out)
// Same tiled skeleton as k_qkv, inputs already bf16.
// A tile is a pure 16B copy -> use GLOBAL_LOAD_ASYNC_TO_LDS_B128 (ASYNCcnt).
// ---------------------------------------------------------------------------
__global__ __launch_bounds__(256)
void k_final(const unsigned short* __restrict__ Mb,
             const unsigned short* __restrict__ qkvb,
             const float* __restrict__ bout, float* __restrict__ y) {
  __shared__ unsigned short lA[8 * 32 * 16];
  __shared__ unsigned short lB[8 * 32 * 16];
  const int b  = blockIdx.z;
  const int o0 = blockIdx.y * 128;
  const int n0 = blockIdx.x * 128;
  const int t = threadIdx.x, lane = t & 31, w = t >> 5;
  const int wr = w >> 1, wc = w & 1;
  const unsigned short* A  = Mb + (size_t)b * CC * HID;
  const unsigned short* Bq = qkvb + (size_t)b * OQKV * NN;   // q rows 0..511

  v8f acc[2][4];
#pragma unroll
  for (int i = 0; i < 2; ++i)
#pragma unroll
    for (int j = 0; j < 4; ++j) acc[i][j] = 0.f;

  for (int ks = 0; ks < HID; ks += 32) {
    __syncthreads();
    // ---- stage A tile via async global->LDS (16B per lane, fragment order)
#pragma unroll
    for (int p = 0; p < 2; ++p) {
      int r = t + p * 256;
      int m = r >> 2, k0 = (r & 3) * 8;
      const unsigned short* src = A + (size_t)(o0 + m) * HID + ks + k0;
      if (ks + 32 < HID) __builtin_prefetch(src + 32, 0, 3);
      int ln = (m & 15) | ((k0 & 8) ? 16 : 0);
      unsigned short* dst = &lA[(((m >> 4) * 32 + ln) * 16) + ((k0 & 16) ? 8 : 0)];
      async_g2l_b128((unsigned)(size_t)(void*)dst, src);
    }
    // ---- stage B tile: strided bf16 gather (must go through VGPRs)
#pragma unroll
    for (int p = 0; p < 2; ++p) {
      int r = t + p * 256;
      int n = r & 127, k0 = (r >> 7) * 8;
      const unsigned short* src = Bq + (size_t)(ks + k0) * NN + n0 + n;
      unsigned short s[8];
#pragma unroll
      for (int i = 0; i < 8; ++i) s[i] = src[(size_t)i * NN];
      U4 o;
      o.x = (unsigned)s[0] | ((unsigned)s[1] << 16);
      o.y = (unsigned)s[2] | ((unsigned)s[3] << 16);
      o.z = (unsigned)s[4] | ((unsigned)s[5] << 16);
      o.w = (unsigned)s[6] | ((unsigned)s[7] << 16);
      int ln = (n & 15) | ((k0 & 8) ? 16 : 0);
      *(U4*)&lB[(((n >> 4) * 32 + ln) * 16) + ((k0 & 16) ? 8 : 0)] = o;
    }
    wait_asynccnt0();        // A-tile async copies complete before barrier
    __syncthreads();
    Frag af[2];
#pragma unroll
    for (int mi = 0; mi < 2; ++mi) {
      const U4* ap = (const U4*)&lA[((wr * 2 + mi) * 32 + lane) * 16];
      af[mi].q[0] = ap[0]; af[mi].q[1] = ap[1];
    }
#pragma unroll
    for (int ni = 0; ni < 4; ++ni) {
      Frag bf_;
      const U4* bp = (const U4*)&lB[((wc * 4 + ni) * 32 + lane) * 16];
      bf_.q[0] = bp[0]; bf_.q[1] = bp[1];
#pragma unroll
      for (int mi = 0; mi < 2; ++mi) acc[mi][ni] = wmma_bf16(af[mi], bf_, acc[mi][ni]);
    }
  }
#pragma unroll
  for (int mi = 0; mi < 2; ++mi)
#pragma unroll
    for (int ni = 0; ni < 4; ++ni) {
      int ncol = n0 + wc * 64 + ni * 16 + (lane & 15);
#pragma unroll
      for (int r = 0; r < 8; ++r) {
        int orow = o0 + wr * 32 + mi * 16 + r + ((lane >> 4) * 8);
        y[((size_t)b * CC + orow) * NN + ncol] = acc[mi][ni][r] + bout[orow];
      }
    }
}

// ---------------------------------------------------------------------------
// Host launcher
// ---------------------------------------------------------------------------
extern "C" void kernel_launch(void* const* d_in, const int* in_sizes, int n_in,
                              void* d_out, int out_size, void* d_ws, size_t ws_size,
                              hipStream_t stream) {
  const float* x     = (const float*)d_in[0];
  const float* wqkv  = (const float*)d_in[1];
  const float* wout  = (const float*)d_in[2];
  const float* bout  = (const float*)d_in[3];
  float*       y     = (float*)d_out;

  char* ws = (char*)d_ws;
  unsigned short* qkvb = (unsigned short*)ws;                 // 16*1536*4096 bf16 = 192 MiB
  size_t off = (size_t)BB * OQKV * NN * sizeof(unsigned short);
  float* rowmax = (float*)(ws + off); off += (size_t)BB * HID * sizeof(float);
  float* rowsum = (float*)(ws + off); off += (size_t)BB * HID * sizeof(float);
  float* ctx    = (float*)(ws + off); off += (size_t)BB * NH * DH * DH * sizeof(float);
  unsigned short* Mb = (unsigned short*)(ws + off);           // 16*256*512 bf16 = 4 MiB

  dim3 g1(NN / 128, OQKV / 128, BB);        // 32 x 12 x 16
  k_qkv<<<g1, 256, 0, stream>>>(x, wqkv, qkvb);

  k_stats<<<BB * HID, 256, 0, stream>>>(qkvb, rowmax, rowsum);

  k_ctx<<<BB * NH, 128, 0, stream>>>(qkvb, rowmax, rowsum, ctx);

  dim3 g4(NH, BB);                          // 8 x 16
  k_fold<<<g4, 256, 0, stream>>>(wout, ctx, Mb);

  dim3 g5(NN / 128, CC / 128, BB);          // 32 x 2 x 16
  k_final<<<g5, 256, 0, stream>>>(Mb, qkvb, bout, y);
}
